// PolynomialLayer_17927193494221
// MI455X (gfx1250) — matrix operand order repaired and checked by reading, not compiled, not run on hardware
//
#include <hip/hip_runtime.h>
#include <hip/hip_bf16.h>
#include <math.h>

typedef __attribute__((ext_vector_type(16))) _Float16 v16h;
typedef __attribute__((ext_vector_type(8)))  float    v8f;

#define NDEG 9      // POLY_DEGREE + 1
#define WCH  64     // output channels

// Hardware tanh (CDNA5 V_TANH_F32, TRANS unit) instead of branchy OCML tanhf.
__device__ __forceinline__ float hw_tanh(float v) {
#if __has_builtin(__builtin_amdgcn_tanhf)
  return __builtin_amdgcn_tanhf(v);
#else
  float r;
  // TRANS ops need 1 independent op before the result is consumed; the
  // compiler can't see through asm, so pad with a v_nop ourselves.
  asm volatile("v_tanh_f32 %0, %1\n\tv_nop" : "=v"(r) : "v"(v));
  return r;
#endif
}

// ---------------------------------------------------------------------------
// Prep kernel: pack [Wx ; Wt] (K=18, padded to 32) x 64 channels into the
// WMMA B-matrix VGPR layout (32x16 f16 per 16-channel tile, 4 tiles).
// Contiguous k order (as A, not the ISA table order): lanes 0-15 hold K=0..15 (N=lane),
// lanes 16-31 hold K=16..31 (N=lane-16); h-th half in the lane's v16h = K%16.
// ws element index e = ((t*32 + lane)*16 + h).
// ---------------------------------------------------------------------------
__global__ void poly_pack_b(const float* __restrict__ wx,
                            const float* __restrict__ wt,
                            _Float16* __restrict__ pb) {
  int e = blockIdx.x * blockDim.x + threadIdx.x;
  if (e >= 4 * 32 * 16) return;
  int t    = e >> 9;
  int lane = (e >> 4) & 31;
  int h    = e & 15;
  int n = lane & 15;
  int g = lane >> 4;
  int K = g * 16 + h;          // reduction index 0..31
  int m = t * 16 + n;          // output channel 0..63
  float v = 0.f;
  if (K < NDEG)          v = wx[m * NDEG + K];
  else if (K < 2 * NDEG) v = wt[m * NDEG + (K - NDEG)];
  pb[e] = (_Float16)v;
}

// ---------------------------------------------------------------------------
// Main kernel: one wave handles 16 flattened positions per tile iteration.
// Memory-bound streaming (x read 256MB + out write 256MB @ 23.3TB/s ~ 22us);
// WMMA does the [16x18] @ [18x64] basis projection; V_TANH_F32 epilogue.
// ---------------------------------------------------------------------------
__global__ __launch_bounds__(256)
void poly_wmma_kernel(const float* __restrict__ x,
                      const float* __restrict__ xg,
                      const float* __restrict__ tg,
                      const _Float16* __restrict__ pb,
                      const float* __restrict__ bias,
                      float* __restrict__ out,
                      int ntiles) {
  const int lane   = threadIdx.x & 31;
  const int wave   = (blockIdx.x * blockDim.x + threadIdx.x) >> 5;
  const int nwaves = (gridDim.x * blockDim.x) >> 5;
  const int n = lane & 15;     // row within lane-group / channel within tile
  const int g = lane >> 4;     // lane group (0 or 1)

  // B tiles: persistent across the grid-stride loop (4 x v16h = 32 VGPRs).
  const v16h b0 = *(const v16h*)(pb + (0 * 32 + lane) * 16);
  const v16h b1 = *(const v16h*)(pb + (1 * 32 + lane) * 16);
  const v16h b2 = *(const v16h*)(pb + (2 * 32 + lane) * 16);
  const v16h b3 = *(const v16h*)(pb + (3 * 32 + lane) * 16);

  // bias per (tile t, channel n) is loop-invariant.
  float bm0 = bias[0 * 16 + n];
  float bm1 = bias[1 * 16 + n];
  float bm2 = bias[2 * 16 + n];
  float bm3 = bias[3 * 16 + n];

  for (int T = wave; T < ntiles; T += nwaves) {
    const int base = T * 16;

    // Prefetch next tile's x block: 32 lanes x 128B lines = the full 4KB tile.
    if (T + nwaves < ntiles) {
      __builtin_prefetch(&x[(size_t)(T + nwaves) * 16 * WCH + lane * 32], 0, 1);
    }

    // --- Legendre bases for this lane's row (position base + n) ---
    const float xv = xg[base + n];
    const float tv = tg[base + n];
    float Px[NDEG], Pt[NDEG];
    Px[0] = 1.f; Px[1] = xv;
    Pt[0] = 1.f; Pt[1] = tv;
#pragma unroll
    for (int k = 2; k < NDEG; ++k) {
      const float c1 = (float)(2 * k - 1) / (float)k;
      const float c2 = (float)(k - 1) / (float)k;
      Px[k] = c1 * xv * Px[k - 1] - c2 * Px[k - 2];
      Pt[k] = c1 * tv * Pt[k - 1] - c2 * Pt[k - 2];
    }

    // --- Pack A (16x32 f16): group 0 -> K in {0..15},
    //                         group 1 -> K in {16..31} ---
    v16h a;
#pragma unroll
    for (int h = 0; h < 16; ++h) {
      const int K0 = h;         // lane group 0
      const int K1 = 16 + h;    // lane group 1
      const float v0 = (K0 < NDEG) ? Px[K0] : ((K0 < 2 * NDEG) ? Pt[K0 - NDEG] : 0.f);
      const float v1 = (K1 < NDEG) ? Px[K1] : ((K1 < 2 * NDEG) ? Pt[K1 - NDEG] : 0.f);
      a[h] = (_Float16)(g ? v1 : v0);
    }

    // --- 4 WMMAs cover all 64 channels; fused, branch-free epilogue ---
#pragma unroll
    for (int t = 0; t < 4; ++t) {
      const v16h b   = (t == 0) ? b0 : (t == 1) ? b1 : (t == 2) ? b2 : b3;
      const float bm = (t == 0) ? bm0 : (t == 1) ? bm1 : (t == 2) ? bm2 : bm3;
      v8f c = {};
      c = __builtin_amdgcn_wmma_f32_16x16x32_f16(
          /*neg_a=*/false, a, /*neg_b=*/false, b,
          /*c_mod=*/(short)0, c, /*reuse_a=*/false, /*reuse_b=*/false);
      const int m = t * 16 + n;
#pragma unroll
      for (int r = 0; r < 8; ++r) {
        const int p   = base + g * 8 + r;   // D-matrix row M = g*8 + r
        const int idx = p * WCH + m;        // max 67M elements: fits int32
        out[idx] = hw_tanh(fmaf(x[idx], c[r], bm));
      }
    }
  }
}

// ---------------------------------------------------------------------------
// Host launcher. Input order: x, x_grid, t_grid, weights_x, weights_t, bias.
// ---------------------------------------------------------------------------
extern "C" void kernel_launch(void* const* d_in, const int* in_sizes, int n_in,
                              void* d_out, int out_size, void* d_ws, size_t ws_size,
                              hipStream_t stream) {
  const float* x    = (const float*)d_in[0];
  const float* xg   = (const float*)d_in[1];
  const float* tg   = (const float*)d_in[2];
  const float* wx   = (const float*)d_in[3];
  const float* wt   = (const float*)d_in[4];
  const float* bias = (const float*)d_in[5];
  float* out = (float*)d_out;
  _Float16* pb = (_Float16*)d_ws;   // 4*32*16 halves = 4 KB

  const int npos   = in_sizes[1];   // B*I*J (x_grid element count)
  const int ntiles = npos / 16;     // 65536 for the reference shapes

  hipLaunchKernelGGL(poly_pack_b, dim3(8), dim3(256), 0, stream, wx, wt, pb);

  // 2048 blocks x 8 waves = 16384 waves -> 4 tiles/wave: enough memory-level
  // parallelism to saturate HBM while amortizing the B-tile / bias loads.
  int blocks = 2048;
  int max_blocks = (ntiles + 7) / 8;          // >= 1 wave-tile per wave
  if (blocks > max_blocks) blocks = max_blocks;
  if (blocks < 1) blocks = 1;

  hipLaunchKernelGGL(poly_wmma_kernel, dim3(blocks), dim3(256), 0, stream,
                     x, xg, tg, pb, bias, out, ntiles);
}
